// SPECausalReasoner_80985903333701
// MI455X (gfx1250) — matrix operand order repaired and checked by reading, not compiled, not run on hardware
//
#include <hip/hip_runtime.h>
#include <hip/hip_bf16.h>
#include <math.h>

typedef __attribute__((ext_vector_type(16))) _Float16 v16h;
typedef __attribute__((ext_vector_type(8)))  _Float16 v8h;
typedef __attribute__((ext_vector_type(8)))  float    v8f;

#define WAVES_PER_BLOCK 8

static __device__ __forceinline__ v16h cat16(v8h lo, v8h hi){
  return __builtin_shufflevector(lo, hi, 0,1,2,3,4,5,6,7,8,9,10,11,12,13,14,15);
}
static __device__ __forceinline__ v8f wmma_f16(v16h a, v16h b, v8f c){
  // D = A(16x32 f16) * B(32x16 f16) + C(16x16 f32)
  return __builtin_amdgcn_wmma_f32_16x16x32_f16(false, a, false, b, (short)0, c, false, false);
}
// Branch-free tanh-form GELU on native transcendentals (v_exp_f32 / v_rcp_f32)
static __device__ __forceinline__ float gelu_fast(float x){
  float t = x * x;
  float p = __builtin_fmaf(0.044715f * t, x, x);
  float z = __builtin_amdgcn_exp2f(-2.302118131f * p);
  return x * __builtin_amdgcn_rcpf(1.0f + z);
}
static __device__ __forceinline__ float sigmoid_fast(float x){
  float z = __builtin_amdgcn_exp2f(-1.4426950408889634f * x);
  return __builtin_amdgcn_rcpf(1.0f + z);
}
// A operand (16x32 f16 tile) from row-major f16 LDS staging [16][Kd]
static __device__ __forceinline__ v16h loadA(const _Float16* base, int Kd, int m, int half, int kt){
  const _Float16* p = base + m*Kd + kt*32 + half*8;
  v8h lo = *(const v8h*)p;
  v8h hi = *(const v8h*)(p + 16);
  return cat16(lo, hi);
}
// B operand (32x16 f16 tile) from column-major f16 LDS weights [N][Kd]
static __device__ __forceinline__ v16h loadB(const _Float16* base, int Kd, int ncolTot, int half, int kt){
  const _Float16* p = base + ncolTot*Kd + kt*32 + half*8;
  v8h lo = *(const v8h*)p;
  v8h hi = *(const v8h*)(p + 16);
  return cat16(lo, hi);
}

// ---------------- level-bucketing (counting sort -> permutation in d_ws) ----------------
// ws int layout: [0..2] counts, [4..6] allocation cursors, [8..] perm (Nrows ints)
__global__ void zero_ws_kernel(int* w){
  if (threadIdx.x < 8) w[threadIdx.x] = 0;
}
__global__ __launch_bounds__(256) void count_levels_kernel(const int* __restrict__ lv, int n, int* __restrict__ w){
  __shared__ int c[3];
  if (threadIdx.x < 3) c[threadIdx.x] = 0;
  __syncthreads();
  int i = blockIdx.x * 256 + threadIdx.x;
  if (i < n) atomicAdd(&c[lv[i]], 1);
  __syncthreads();
  if (threadIdx.x < 3) atomicAdd(&w[threadIdx.x], c[threadIdx.x]);
}
__global__ void make_bases_kernel(int* w){
  if (threadIdx.x == 0){ w[4] = 0; w[5] = w[0]; w[6] = w[0] + w[1]; }
}
__global__ __launch_bounds__(256) void scatter_perm_kernel(const int* __restrict__ lv, int n,
                                                           int* __restrict__ w, int* __restrict__ perm){
  __shared__ int c[3], base[3];
  if (threadIdx.x < 3) c[threadIdx.x] = 0;
  __syncthreads();
  int i = blockIdx.x * 256 + threadIdx.x;
  int l = 0, r = 0;
  if (i < n){ l = lv[i]; r = atomicAdd(&c[l], 1); }
  __syncthreads();
  if (threadIdx.x < 3) base[threadIdx.x] = atomicAdd(&w[4 + threadIdx.x], c[threadIdx.x]);
  __syncthreads();
  if (i < n) perm[base[l] + r] = i;
}

// ---------------- main fused-MLP kernel ----------------
__global__ __launch_bounds__(256) void reasoner_kernel(
    const float* __restrict__ feats, const int* __restrict__ levels,
    const float* __restrict__ g_w1, const float* __restrict__ g_b1,
    const float* __restrict__ g_w2, const float* __restrict__ g_b2,
    const float* __restrict__ o_w1, const float* __restrict__ o_b1,
    const float* __restrict__ o_w2, const float* __restrict__ o_b2,
    const float* __restrict__ h_w1, const float* __restrict__ h_b1,
    const float* __restrict__ h_w2, const float* __restrict__ h_b2,
    const float* __restrict__ c_w1, const float* __restrict__ c_b1,
    const float* __restrict__ c_w2, const float* __restrict__ c_b2,
    float* __restrict__ out, const int* __restrict__ perm, int use_perm, int ntiles)
{
  __shared__ __align__(16) _Float16 sW1t[64*32];
  __shared__ __align__(16) _Float16 sW2t[128*64];
  __shared__ __align__(16) _Float16 sHt1[3*64*128];
  __shared__ __align__(16) _Float16 sHt2[3*32*64];
  __shared__ __align__(16) _Float16 sCt1[16*32];
  __shared__ float sB1[64], sB2[128], sBH1[192], sBH2[96], sBC1[16], sCW2[16];
  __shared__ __align__(16) _Float16 sH1[WAVES_PER_BLOCK][16*64];
  __shared__ __align__(16) _Float16 sX [WAVES_PER_BLOCK][16*128];
  __shared__ __align__(16) _Float16 sO [WAVES_PER_BLOCK][16*32];

  const int tid = threadIdx.x;

  // ---- stage + transpose + f32->f16 convert weights (once per block) ----
  for (int i = tid; i < 64*32; i += 256){
    int nn = i >> 5, k = i & 31; float v = 0.f;
    if (nn < 32) { if (k < 5)            v = g_w1[k*32 + nn]; }
    else         { if (k >= 5 && k < 10) v = o_w1[(k-5)*32 + (nn-32)]; }
    sW1t[i] = (_Float16)v;
  }
  for (int i = tid; i < 128*64; i += 256){
    int nn = i >> 6, k = i & 63; float v = 0.f;
    if (nn < 64) { if (k < 32)  v = g_w2[k*64 + nn]; }
    else         { if (k >= 32) v = o_w2[(k-32)*64 + (nn-64)]; }
    sW2t[i] = (_Float16)v;
  }
  for (int i = tid; i < 3*64*128; i += 256){
    int l = i >> 13, r = i & 8191, nn = r >> 7, k = r & 127;
    sHt1[i] = (_Float16)h_w1[l*8192 + k*64 + nn];
  }
  for (int i = tid; i < 3*32*64; i += 256){
    int l = i >> 11, r = i & 2047, nn = r >> 6, k = r & 63;
    sHt2[i] = (_Float16)h_w2[l*2048 + k*32 + nn];
  }
  for (int i = tid; i < 16*32; i += 256){
    int nn = i >> 5, k = i & 31;
    sCt1[i] = (_Float16)c_w1[k*16 + nn];
  }
  if (tid < 64)  sB1[tid]  = (tid < 32) ? g_b1[tid] : o_b1[tid-32];
  if (tid < 128) sB2[tid]  = (tid < 64) ? g_b2[tid] : o_b2[tid-64];
  if (tid < 192) sBH1[tid] = h_b1[tid];
  if (tid < 96)  sBH2[tid] = h_b2[tid];
  if (tid < 16)  { sBC1[tid] = c_b1[tid]; sCW2[tid] = c_w2[tid]; }
  __syncthreads();

  const int wave = tid >> 5;
  const int lane = tid & 31;
  const int ncol = lane & 15;
  const int half = lane >> 4;
  const float cb2 = c_b2[0];

  _Float16* h1s = sH1[wave];
  _Float16* xs  = sX[wave];
  _Float16* os  = sO[wave];

  const int tstep = gridDim.x * WAVES_PER_BLOCK;
  for (int t = blockIdx.x * WAVES_PER_BLOCK + wave; t < ntiles; t += tstep)
  {
    const int rowbase = t << 4;
    const int* pbase = perm + rowbase;

    // levels for routing; C-layout rows M = v + 8*half (gathered through perm when sorted)
    int lvrow[8];
    #pragma unroll
    for (int v = 0; v < 8; ++v){
      int idx = v + 8*half;
      int r = use_perm ? pbase[idx] : (rowbase + idx);
      lvrow[v] = levels[r];
    }

    // ---- build A0 (16x32 f16, K=0..9 valid) straight from global, in WMMA A layout ----
    int ra = use_perm ? pbase[ncol] : (rowbase + ncol);
    v16h a0;
    #pragma unroll
    for (int i = 0; i < 16; ++i) a0[i] = (_Float16)0.0f;
    const float* frow = feats + (size_t)ra * 10;
    if (half == 0){
      #pragma unroll
      for (int i = 0; i < 8; ++i) a0[i] = (_Float16)frow[i];   // K = 0..7
    } else {
      a0[0] = (_Float16)frow[8];                               // K = 8
      a0[1] = (_Float16)frow[9];                               // K = 9
    }

    // ---- fused encoders, layer 1: [16x32] @ blockdiag[32x64] -> h1 [16x64] ----
    #pragma unroll
    for (int nt = 0; nt < 4; ++nt){
      float b = sB1[nt*16 + ncol];
      v8f c = {b,b,b,b,b,b,b,b};
      v16h bm = loadB(sW1t, 32, nt*16 + ncol, half, 0);
      c = wmma_f16(a0, bm, c);
      #pragma unroll
      for (int v = 0; v < 8; ++v)
        h1s[(v + 8*half)*64 + nt*16 + ncol] = (_Float16)gelu_fast(c[v]);
    }
    // ---- fused encoders, layer 2: [16x64] @ blockdiag[64x128] -> x [16x128] ----
    #pragma unroll
    for (int nt = 0; nt < 8; ++nt){
      float b = sB2[nt*16 + ncol];
      v8f c = {b,b,b,b,b,b,b,b};
      #pragma unroll
      for (int kt = 0; kt < 2; ++kt){
        v16h a  = loadA(h1s, 64, ncol, half, kt);
        v16h bm = loadB(sW2t, 64, nt*16 + ncol, half, kt);
        c = wmma_f16(a, bm, c);
      }
      #pragma unroll
      for (int v = 0; v < 8; ++v)
        xs[(v + 8*half)*128 + nt*16 + ncol] = (_Float16)gelu_fast(c[v]);
    }
    // ---- heads (128->64->32): skip heads not present in this tile (wave-uniform branch).
    //      With level-sorted permutation, all but ~2 boundary tiles run exactly ONE head.
    float oa[2][8];
    #pragma unroll
    for (int q = 0; q < 2; ++q)
      #pragma unroll
      for (int v = 0; v < 8; ++v) oa[q][v] = 0.f;
    for (int l = 0; l < 3; ++l){
      bool need = false;
      #pragma unroll
      for (int v = 0; v < 8; ++v) need |= (lvrow[v] == l);
      if (__ballot(need) == 0ull) continue;   // scalar branch; EXEC stays all-ones
      const _Float16* w1t = sHt1 + l*64*128;
      const _Float16* w2t = sHt2 + l*32*64;
      #pragma unroll
      for (int nt = 0; nt < 4; ++nt){
        float b = sBH1[l*64 + nt*16 + ncol];
        v8f c = {b,b,b,b,b,b,b,b};
        #pragma unroll
        for (int kt = 0; kt < 4; ++kt){
          v16h a  = loadA(xs, 128, ncol, half, kt);
          v16h bm = loadB(w1t, 128, nt*16 + ncol, half, kt);
          c = wmma_f16(a, bm, c);
        }
        #pragma unroll
        for (int v = 0; v < 8; ++v)
          h1s[(v + 8*half)*64 + nt*16 + ncol] = (_Float16)gelu_fast(c[v]);
      }
      #pragma unroll
      for (int nt = 0; nt < 2; ++nt){
        float b = sBH2[l*32 + nt*16 + ncol];
        v8f c = {b,b,b,b,b,b,b,b};
        #pragma unroll
        for (int kt = 0; kt < 2; ++kt){
          v16h a  = loadA(h1s, 64, ncol, half, kt);
          v16h bm = loadB(w2t, 64, nt*16 + ncol, half, kt);
          c = wmma_f16(a, bm, c);
        }
        #pragma unroll
        for (int v = 0; v < 8; ++v)
          if (lvrow[v] == l) oa[nt][v] = gelu_fast(c[v]);
      }
    }
    #pragma unroll
    for (int nt = 0; nt < 2; ++nt)
      #pragma unroll
      for (int v = 0; v < 8; ++v)
        os[(v + 8*half)*32 + nt*16 + ncol] = (_Float16)oa[nt][v];

    // ---- classifier: 32->16 via WMMA, 16->1 via cross-lane reduction ----
    {
      float b = sBC1[ncol];
      v8f c = {b,b,b,b,b,b,b,b};
      v16h a  = loadA(os, 32, ncol, half, 0);
      v16h bm = loadB(sCt1, 32, ncol, half, 0);
      c = wmma_f16(a, bm, c);
      const float w2 = sCW2[ncol];
      float part[8];
      #pragma unroll
      for (int v = 0; v < 8; ++v) part[v] = gelu_fast(c[v]) * w2;
      #pragma unroll
      for (int msk = 1; msk <= 8; msk <<= 1)
        #pragma unroll
        for (int v = 0; v < 8; ++v) part[v] += __shfl_xor(part[v], msk, 32);
      if (ncol < 8){
        int oidx = rowbase + 8*half + ncol;
        int orow = use_perm ? perm[oidx] : oidx;
        out[orow] = sigmoid_fast(part[ncol] + cb2);
      }
    }
  }
}

extern "C" void kernel_launch(void* const* d_in, const int* in_sizes, int n_in,
                              void* d_out, int out_size, void* d_ws, size_t ws_size,
                              hipStream_t stream) {
  const float* feats  = (const float*)d_in[0];
  const int*   levels = (const int*)  d_in[1];
  const float* g_w1 = (const float*)d_in[2];
  const float* g_b1 = (const float*)d_in[3];
  const float* g_w2 = (const float*)d_in[4];
  const float* g_b2 = (const float*)d_in[5];
  const float* o_w1 = (const float*)d_in[6];
  const float* o_b1 = (const float*)d_in[7];
  const float* o_w2 = (const float*)d_in[8];
  const float* o_b2 = (const float*)d_in[9];
  const float* h_w1 = (const float*)d_in[10];
  const float* h_b1 = (const float*)d_in[11];
  const float* h_w2 = (const float*)d_in[12];
  const float* h_b2 = (const float*)d_in[13];
  const float* c_w1 = (const float*)d_in[14];
  const float* c_b1 = (const float*)d_in[15];
  const float* c_w2 = (const float*)d_in[16];
  const float* c_b2 = (const float*)d_in[17];
  float* out = (float*)d_out;

  const int Nrows  = in_sizes[0] / 10;
  const int ntiles = Nrows >> 4;           // N = 1,048,576 -> 65,536 tiles of 16 rows

  int* w = (int*)d_ws;
  int* perm = w + 8;
  const size_t ws_needed = (size_t)(Nrows + 8) * sizeof(int);
  const int use_perm = (ws_size >= ws_needed) ? 1 : 0;

  if (use_perm){
    const int nblk = (Nrows + 255) / 256;
    zero_ws_kernel<<<1, 8, 0, stream>>>(w);
    count_levels_kernel<<<nblk, 256, 0, stream>>>(levels, Nrows, w);
    make_bases_kernel<<<1, 1, 0, stream>>>(w);
    scatter_perm_kernel<<<nblk, 256, 0, stream>>>(levels, Nrows, w, perm);
  }

  dim3 grid(2048), block(256);             // 8 waves/block, grid-stride: 4 tiles per wave
  reasoner_kernel<<<grid, block, 0, stream>>>(
      feats, levels, g_w1, g_b1, g_w2, g_b2, o_w1, o_b1, o_w2, o_b2,
      h_w1, h_b1, h_w2, h_b2, c_w1, c_b1, c_w2, c_b2, out, perm, use_perm, ntiles);
}
